// MultiHeadAttention_34187939676838
// MI455X (gfx1250) — matrix operand order repaired and checked by reading, not compiled, not run on hardware
//
#include <hip/hip_runtime.h>
#include <hip/hip_bf16.h>

typedef __attribute__((ext_vector_type(16))) __bf16 v16bf;
typedef __attribute__((ext_vector_type(8)))  __bf16 v8bf;
typedef __attribute__((ext_vector_type(4)))  __bf16 v4bf;
typedef __attribute__((ext_vector_type(8)))  float  v8f;

// TDM descriptor group types (per probe: g0=uint32x4, g1=int32x8, g2/g3=int32x4)
typedef __attribute__((ext_vector_type(4))) unsigned int tdm_v4u;
typedef __attribute__((ext_vector_type(8))) int          tdm_v8i;
typedef __attribute__((ext_vector_type(4))) int          tdm_v4i;

#if defined(__HIP_DEVICE_COMPILE__) && __has_builtin(__builtin_amdgcn_tensor_load_to_lds)
#define USE_TDM 1
#else
#define USE_TDM 0
#endif

#define DEV static __device__ __forceinline__

constexpr int Bc = 2, Sc = 2048, Dc = 1024, Hc = 16, DKc = 64;

DEV __bf16 to_bf16(float f) {
  unsigned u = __builtin_bit_cast(unsigned, f);
  u += 0x7fffu + ((u >> 16) & 1u);               // round-to-nearest-even
  unsigned short h = (unsigned short)(u >> 16);
  return __builtin_bit_cast(__bf16, h);
}

// A-matrix fragment 16x32 (MxK), row-major source with row stride `stride`
// (elements) starting at column kOff. ISA: lanes 0-15 -> M=lane, K 0..7 in
// v0-3 / 16..23 in v4-7; lanes 16-31 -> same M, K 8..15 / 24..31.
DEV v16bf load_frag_a(const __bf16* p, int stride, int kOff) {
  const int lane = threadIdx.x & 31;
  const int m    = lane & 15;
  const int koff = (lane >> 4) * 8;
  const __bf16* row = p + (size_t)m * stride + kOff;
  v8bf lo = *(const v8bf*)(row + koff);
  v8bf hi = *(const v8bf*)(row + 16 + koff);
  v16bf r;
#pragma unroll
  for (int i = 0; i < 8; ++i) { r[i] = lo[i]; r[8 + i] = hi[i]; }
  return r;
}

// B-matrix fragment 32x16 (KxN), source stored N-major: B[k][n] =
// p[n*stride + kOff + k]. ISA: lane n = lane&15; lanes 0-15 K 0..7 / 16..23,
// lanes 16-31 K 8..15 / 24..31, packed 2 per VGPR.
DEV v16bf load_frag_b(const __bf16* p, int stride, int kOff) {
  const int lane = threadIdx.x & 31;
  const int n    = lane & 15;
  const int kb   = (lane >> 4) * 8;
  const __bf16* row = p + (size_t)n * stride + kOff + kb;
  v8bf lo = *(const v8bf*)(row);
  v8bf hi = *(const v8bf*)(row + 16);
  v16bf r;
#pragma unroll
  for (int i = 0; i < 8; ++i) { r[i] = lo[i]; r[8 + i] = hi[i]; }
  return r;
}

DEV v8f wmma_bf16(v16bf a, v16bf b, v8f c) {
  return __builtin_amdgcn_wmma_f32_16x16x32_bf16(false, a, false, b,
                                                 (short)0, c, false, false);
}

// Y = X @ W^T + bias.  X: [M,K] (float or bf16), W: [N,K] fp32.
// OUT_MODE: 0 = fp32 row-major [M,N]; 1 = bf16 [B,H,S,DK]; 2 = bf16 [B,H,DK,S].
template <typename XT, int OUT_MODE>
__global__ __launch_bounds__(128) void gemm_wmma_kernel(
    const XT* __restrict__ X, const float* __restrict__ W,
    const float* __restrict__ bias, void* __restrict__ Yv,
    int M, int N, int K) {
  __shared__ alignas(16) __bf16 As[128][40];   // BM=128, BK=32 (+pad)
  __shared__ alignas(16) __bf16 Bs[64][40];    // BN=64

  const int tid  = threadIdx.x;
  const int wave = tid >> 5;
  const int lane = tid & 31;
  const int m0 = blockIdx.y * 128;
  const int n0 = blockIdx.x * 64;

  const v8f zero = {0.f, 0.f, 0.f, 0.f, 0.f, 0.f, 0.f, 0.f};
  v8f acc[2][4];
#pragma unroll
  for (int i = 0; i < 2; ++i)
#pragma unroll
    for (int j = 0; j < 4; ++j) acc[i][j] = zero;

  for (int k0 = 0; k0 < K; k0 += 32) {
    __syncthreads();
#pragma unroll
    for (int it = 0; it < 8; ++it) {           // A tile: 128x32
      int idx = it * 128 + tid;
      int r = idx >> 3;
      int c = (idx & 7) * 4;
      const XT* xa = X + (size_t)(m0 + r) * K + k0 + c;
      if constexpr (sizeof(XT) == 4) {
        float4 xv = *(const float4*)xa;
        As[r][c + 0] = to_bf16(xv.x); As[r][c + 1] = to_bf16(xv.y);
        As[r][c + 2] = to_bf16(xv.z); As[r][c + 3] = to_bf16(xv.w);
      } else {
        v4bf xv = *(const v4bf*)xa;
        As[r][c + 0] = xv[0]; As[r][c + 1] = xv[1];
        As[r][c + 2] = xv[2]; As[r][c + 3] = xv[3];
      }
    }
#pragma unroll
    for (int it = 0; it < 4; ++it) {           // B tile: 64x32
      int idx = it * 128 + tid;
      int r = idx >> 3;
      int c = (idx & 7) * 4;
      float4 wv = *(const float4*)(W + (size_t)(n0 + r) * K + k0 + c);
      Bs[r][c + 0] = to_bf16(wv.x); Bs[r][c + 1] = to_bf16(wv.y);
      Bs[r][c + 2] = to_bf16(wv.z); Bs[r][c + 3] = to_bf16(wv.w);
    }
    if (k0 + 32 < K) {                         // hint next K-tile into caches
      int r0 = tid >> 3, c0 = (tid & 7) * 4;
      __builtin_prefetch(X + (size_t)(m0 + r0) * K + k0 + 32 + c0, 0, 1);
      __builtin_prefetch(W + (size_t)(n0 + r0) * K + k0 + 32 + c0, 0, 1);
    }
    __syncthreads();

    v16bf a0 = load_frag_a(&As[wave * 32][0], 40, 0);
    v16bf a1 = load_frag_a(&As[wave * 32 + 16][0], 40, 0);
#pragma unroll
    for (int nt = 0; nt < 4; ++nt) {
      v16bf bb = load_frag_b(&Bs[nt * 16][0], 40, 0);
      acc[0][nt] = wmma_bf16(a0, bb, acc[0][nt]);
      acc[1][nt] = wmma_bf16(a1, bb, acc[1][nt]);
    }
  }

  const int rbase = (lane >> 4) * 8;   // C layout: lanes>=16 hold rows 8..15
  const int cc    = lane & 15;
#pragma unroll
  for (int mt = 0; mt < 2; ++mt)
#pragma unroll
    for (int nt = 0; nt < 4; ++nt)
#pragma unroll
      for (int r = 0; r < 8; ++r) {
        int gm = m0 + wave * 32 + mt * 16 + rbase + r;
        int gn = n0 + nt * 16 + cc;
        float y = acc[mt][nt][r] + bias[gn];
        int b = gm >> 11, s = gm & (Sc - 1);
        int h = gn >> 6,  dk = gn & (DKc - 1);
        if constexpr (OUT_MODE == 1) {         // [B,H,S,DK]
          ((__bf16*)Yv)[(((size_t)(b * Hc + h)) * Sc + s) * DKc + dk] = to_bf16(y);
        } else if constexpr (OUT_MODE == 2) {  // [B,H,DK,S]  (for TDM V tiles)
          ((__bf16*)Yv)[(((size_t)(b * Hc + h)) * DKc + dk) * Sc + s] = to_bf16(y);
        } else {
          ((float*)Yv)[(size_t)gm * N + gn] = y;
        }
      }
}

// Flash-attention. Q/K: bf16 [B*H][S][DK]; V: bf16 [B*H][DK][S] (transposed).
// Writes bf16 ctx [B,S,D].
__global__ __launch_bounds__(128) void attn_kernel(
    const __bf16* __restrict__ Qp, const __bf16* __restrict__ Kp,
    const __bf16* __restrict__ VT, __bf16* __restrict__ Ctx) {
  __shared__ alignas(16) __bf16 VsT[64][72];      // V tile [dk][key] (+pad)
  __shared__ alignas(16) __bf16 Pt[4][16][72];    // per-wave P tile [row][key]

  const int tid  = threadIdx.x;
  const int wave = tid >> 5;
  const int lane = tid & 31;
  const int bh = blockIdx.y;
  const int b  = bh >> 4;
  const int h  = bh & 15;
  const __bf16* Q  = Qp + (size_t)bh * Sc * DKc;
  const __bf16* K  = Kp + (size_t)bh * Sc * DKc;
  const __bf16* Vh = VT + (size_t)bh * DKc * Sc;

  const int q0 = blockIdx.x * 64 + wave * 16;

  const v16bf aQ0 = load_frag_a(Q + (size_t)q0 * DKc, DKc, 0);
  const v16bf aQ1 = load_frag_a(Q + (size_t)q0 * DKc, DKc, 32);

  const v8f zero = {0.f, 0.f, 0.f, 0.f, 0.f, 0.f, 0.f, 0.f};
  v8f o[4];
#pragma unroll
  for (int i = 0; i < 4; ++i) o[i] = zero;
  float mrun[8], lrun[8];
#pragma unroll
  for (int r = 0; r < 8; ++r) { mrun[r] = -3.0e38f; lrun[r] = 0.f; }

  const int rbase = (lane >> 4) * 8;
  const int cc    = lane & 15;
#if USE_TDM
  const unsigned ldsVsT = (unsigned)(size_t)(void*)&VsT[0][0];
#endif

  for (int key0 = 0; key0 < Sc; key0 += 64) {
    __syncthreads();  // previous iteration done reading VsT
#if USE_TDM
    if (wave == 0) {
      // D# group0: count=1, lds_addr, 57-bit global addr, type=2 ("image")
      unsigned long long ga = (unsigned long long)(size_t)(const void*)(Vh + key0);
      tdm_v4u g0;
      g0[0] = 1u;                                  // count=1, user descriptor
      g0[1] = ldsVsT;                              // LDS byte address
      g0[2] = (unsigned)ga;                        // global_addr[31:0]
      g0[3] = (unsigned)((ga >> 32) & 0x01ffffffu) | 0x80000000u; // [56:32]|type=2
      // D# group1: data_size=2B, pad 4 dwords every 32 dwords (-> 72-elem rows),
      // tensor_dim0=S (keys, contiguous), tensor_dim1=DK, tile 64x64, stride S.
      tdm_v8i g1;
      g1[0] = (int)((1u << 16) | (1u << 20) | (4u << 22) | (3u << 25));
      g1[1] = (int)(((unsigned)Sc & 0xffffu) << 16);          // tensor_dim0 lo
      g1[2] = (int)(((unsigned)Sc >> 16) | (64u << 16));      // dim0 hi | dim1 lo
      g1[3] = (int)(64u << 16);                               // dim1 hi | tile_dim0
      g1[4] = 64;                                             // tile_dim1 (| tile_dim2=0)
      g1[5] = Sc;                                             // tensor_dim0_stride lo
      g1[6] = 0;                                              // stride hi | dim1_stride lo
      g1[7] = 0;
      tdm_v4i gz = {0, 0, 0, 0};
#if __clang_major__ >= 23
      tdm_v8i gz8 = {0, 0, 0, 0, 0, 0, 0, 0};
      __builtin_amdgcn_tensor_load_to_lds(g0, g1, gz, gz, gz8, 0);
#else
      __builtin_amdgcn_tensor_load_to_lds(g0, g1, gz, gz, 0);
#endif
#if __has_builtin(__builtin_amdgcn_s_wait_tensorcnt)
      __builtin_amdgcn_s_wait_tensorcnt(0);
#else
      asm volatile("s_wait_tensorcnt 0x0" ::: "memory");
#endif
    }
#else
    // Fallback: straight 2D copy (V already transposed in global)
#pragma unroll
    for (int it = 0; it < 4; ++it) {
      int idx = it * 128 + tid;
      int dd  = idx >> 3;
      int k8  = (idx & 7) * 8;
      *(v8bf*)&VsT[dd][k8] = *(const v8bf*)(Vh + (size_t)dd * Sc + key0 + k8);
    }
#endif
    __syncthreads();

    // S = Q @ K^T  (N-dim = key, K-dim = dk; K rows are contiguous K-runs)
    v8f s[4];
#pragma unroll
    for (int nt = 0; nt < 4; ++nt) {
      const __bf16* krow = K + (size_t)(key0 + nt * 16) * DKc;
      v8f t = wmma_bf16(aQ0, load_frag_b(krow, DKc, 0), zero);
      s[nt]  = wmma_bf16(aQ1, load_frag_b(krow, DKc, 32), t);
    }

    // Online softmax (rows live across each 16-lane half-group)
#pragma unroll
    for (int nt = 0; nt < 4; ++nt)
#pragma unroll
      for (int r = 0; r < 8; ++r) s[nt][r] *= 0.125f;   // 1/sqrt(64)

    float mloc[8];
#pragma unroll
    for (int r = 0; r < 8; ++r)
      mloc[r] = fmaxf(fmaxf(s[0][r], s[1][r]), fmaxf(s[2][r], s[3][r]));
#pragma unroll
    for (int off = 1; off < 16; off <<= 1)
#pragma unroll
      for (int r = 0; r < 8; ++r)
        mloc[r] = fmaxf(mloc[r], __shfl_xor(mloc[r], off, 16));

    float alpha[8], rs[8];
#pragma unroll
    for (int r = 0; r < 8; ++r) {
      float mn = fmaxf(mrun[r], mloc[r]);
      alpha[r] = __expf(mrun[r] - mn);
      mrun[r]  = mn;
      rs[r]    = 0.f;
    }
#pragma unroll
    for (int nt = 0; nt < 4; ++nt)
#pragma unroll
      for (int r = 0; r < 8; ++r) {
        float p = __expf(s[nt][r] - mrun[r]);
        s[nt][r] = p;
        rs[r] += p;
      }
#pragma unroll
    for (int off = 1; off < 16; off <<= 1)
#pragma unroll
      for (int r = 0; r < 8; ++r) rs[r] += __shfl_xor(rs[r], off, 16);
#pragma unroll
    for (int r = 0; r < 8; ++r) lrun[r] = lrun[r] * alpha[r] + rs[r];
#pragma unroll
    for (int nt = 0; nt < 4; ++nt)
#pragma unroll
      for (int r = 0; r < 8; ++r) o[nt][r] *= alpha[r];

    // C-layout -> A-layout transpose of P through per-wave LDS
    __bf16* pw = &Pt[wave][0][0];
#pragma unroll
    for (int nt = 0; nt < 4; ++nt)
#pragma unroll
      for (int r = 0; r < 8; ++r)
        pw[(size_t)(rbase + r) * 72 + nt * 16 + cc] = to_bf16(s[nt][r]);
    __syncthreads();

    // O += P @ V
    const __bf16* pr = &Pt[wave][0][0];
    v16bf aP0 = load_frag_a(pr, 72, 0);
    v16bf aP1 = load_frag_a(pr, 72, 32);
#pragma unroll
    for (int nt = 0; nt < 4; ++nt) {
      o[nt] = wmma_bf16(aP0, load_frag_b(&VsT[nt * 16][0], 72, 0),  o[nt]);
      o[nt] = wmma_bf16(aP1, load_frag_b(&VsT[nt * 16][0], 72, 32), o[nt]);
    }
  }

  // Normalize and write ctx[b][s][h*DK+dk] (bf16)
  float inv[8];
#pragma unroll
  for (int r = 0; r < 8; ++r) inv[r] = 1.0f / lrun[r];
#pragma unroll
  for (int nt = 0; nt < 4; ++nt)
#pragma unroll
    for (int r = 0; r < 8; ++r) {
      int srow = q0 + rbase + r;
      int d = h * DKc + nt * 16 + cc;
      Ctx[((size_t)b * Sc + srow) * Dc + d] = to_bf16(o[nt][r] * inv[r]);
    }
}

extern "C" void kernel_launch(void* const* d_in, const int* in_sizes, int n_in,
                              void* d_out, int out_size, void* d_ws,
                              size_t ws_size, hipStream_t stream) {
  const float* q  = (const float*)d_in[0];
  const float* k  = (const float*)d_in[1];
  const float* v  = (const float*)d_in[2];
  const float* Wq = (const float*)d_in[3];
  const float* bq = (const float*)d_in[4];
  const float* Wk = (const float*)d_in[5];
  const float* bk = (const float*)d_in[6];
  const float* Wv = (const float*)d_in[7];
  const float* bv = (const float*)d_in[8];
  const float* Wo = (const float*)d_in[9];
  const float* bo = (const float*)d_in[10];

  const size_t elems = (size_t)Bc * Sc * Dc;   // 4M
  __bf16* Qp  = (__bf16*)d_ws;                 // 8 MB each
  __bf16* Kp  = Qp + elems;
  __bf16* Vp  = Kp + elems;                    // stored transposed [B,H,DK,S]
  __bf16* Ctx = Vp + elems;                    // total 32 MB of d_ws

  dim3 blk(128);
  dim3 gproj(Dc / 64, (Bc * Sc) / 128);        // (16, 32)

  gemm_wmma_kernel<float, 1><<<gproj, blk, 0, stream>>>(
      q, Wq, bq, (void*)Qp, Bc * Sc, Dc, Dc);
  gemm_wmma_kernel<float, 1><<<gproj, blk, 0, stream>>>(
      k, Wk, bk, (void*)Kp, Bc * Sc, Dc, Dc);
  gemm_wmma_kernel<float, 2><<<gproj, blk, 0, stream>>>(
      v, Wv, bv, (void*)Vp, Bc * Sc, Dc, Dc);

  attn_kernel<<<dim3(Sc / 64, Bc * Hc), blk, 0, stream>>>(Qp, Kp, Vp, Ctx);

  gemm_wmma_kernel<__bf16, 0><<<gproj, blk, 0, stream>>>(
      Ctx, Wo, bo, d_out, Bc * Sc, Dc, Dc);
}
